// GNN_GAT_7275674600534
// MI455X (gfx1250) — matrix operand order repaired and checked by reading, not compiled, not run on hardware
//
#include <hip/hip_runtime.h>
#include <hip/hip_bf16.h>
#include <cstdint>

#define HID   128
#define HEADS 4
#define CH    32

typedef __attribute__((ext_vector_type(16))) __bf16 v16bf;
typedef __attribute__((ext_vector_type(8)))  float  v8f;

// ---------- helpers: monotonic float<->int key for atomic segment-max ----------
__device__ __forceinline__ int   fkey(float f) { int i = __float_as_int(f); return i >= 0 ? i : (i ^ 0x7FFFFFFF); }
__device__ __forceinline__ float fdec(int k)   { return k >= 0 ? __int_as_float(k) : __int_as_float(k ^ 0x7FFFFFFF); }

// ---------- node encoder: h = x * enc_w + enc_b (rank-1 outer product) ----------
__global__ void k_encode(const float* __restrict__ x, const float* __restrict__ enc_w,
                         const float* __restrict__ enc_b, float* __restrict__ h, int n) {
  int idx = blockIdx.x * blockDim.x + threadIdx.x;
  if (idx >= n * HID) return;
  int node = idx >> 7, k = idx & 127;
  h[idx] = x[node] * enc_w[k] + enc_b[k];
}

// ---------- per-layer reset of accumulators ----------
__global__ void k_reset(float* __restrict__ acc, int* __restrict__ amax_i,
                        float* __restrict__ denom, int n) {
  int idx = blockIdx.x * blockDim.x + threadIdx.x;
  if (idx < n * HID) acc[idx] = 0.0f;
  if (idx < n * HEADS) { amax_i[idx] = (int)0x80000000; denom[idx] = 0.0f; }
}

// ---------- fold (lin_edge_w, att_edge, eenc) into 8 scalars: a_e = attr*scale[h]+bias[h] ----------
__global__ void k_edge_coeff(const float* __restrict__ lin_edge_w,  // [128][128], layer base
                             const float* __restrict__ att_edge,    // [4][32], layer base
                             const float* __restrict__ eenc_w, const float* __restrict__ eenc_b,
                             float* __restrict__ ae /* scale[4], bias[4] */) {
  __shared__ float red[8][128];
  int k = threadIdx.x;  // 0..127
  float w4[HEADS];
  #pragma unroll
  for (int hh = 0; hh < HEADS; ++hh) {
    float s = 0.0f;
    #pragma unroll
    for (int c = 0; c < CH; ++c) s += lin_edge_w[k * HID + hh * CH + c] * att_edge[hh * CH + c];
    w4[hh] = s;
  }
  float ew = eenc_w[k], eb = eenc_b[k];
  #pragma unroll
  for (int hh = 0; hh < HEADS; ++hh) { red[hh][k] = ew * w4[hh]; red[4 + hh][k] = eb * w4[hh]; }
  __syncthreads();
  for (int off = 64; off > 0; off >>= 1) {
    if (k < off) { for (int j = 0; j < 8; ++j) red[j][k] += red[j][k + off]; }
    __syncthreads();
  }
  if (k < 8) ae[k] = red[k][0];
}

// ---------- node GEMM xs = h @ lin_w via bf16 WMMA, f32 accumulate ----------
// Block: 256 threads = 8 waves. Wave w owns M-strip [blk*128 + w*16, +16) x all 128 cols.
// B (lin_w) staged to LDS in per-lane WMMA layout: K(e,half)=half*16+e, N=lane&15.
__global__ __launch_bounds__(256) void k_gemm_xs(const float* __restrict__ h,
                                                 const float* __restrict__ w,   // [128][128] layer base
                                                 float* __restrict__ xs, int n) {
  __shared__ __align__(32) __bf16 ldsB[4 * 8 * 32 * 16];  // [kc][nt][lane][e] = 32 KB
  int tid = threadIdx.x;
  for (int i = 0; i < 64; ++i) {
    int f    = tid * 64 + i;
    int e    = f & 15;
    int lane = (f >> 4) & 31;
    int nt   = (f >> 9) & 7;
    int kc   = f >> 12;
    int K    = kc * 32 + ((lane >> 4) << 4) + e;   // B layout: lanes0-15 K=0..15, lanes16-31 K=16..31
    int col  = nt * 16 + (lane & 15);
    ldsB[f]  = (__bf16)w[K * HID + col];
  }
  __syncthreads();

  int wv = tid >> 5, lane = tid & 31;
  int half = lane >> 4, lm = lane & 15;
  int row = blockIdx.x * 128 + wv * 16 + lm;
  bool valid = row < n;
  const float* hrow = h + (size_t)row * HID;

  v8f acc[8];
  #pragma unroll
  for (int nt = 0; nt < 8; ++nt)
    #pragma unroll
    for (int r = 0; r < 8; ++r) acc[nt][r] = 0.0f;

  for (int kc = 0; kc < 4; ++kc) {
    v16bf a;
    if (valid) {
      int base = kc * 32 + half * 16;  // A layout: K=half*16+e
      #pragma unroll
      for (int e = 0; e < 8; ++e) a[e] = (__bf16)hrow[base + e];
      #pragma unroll
      for (int e = 0; e < 8; ++e) a[8 + e] = (__bf16)hrow[base + 8 + e];
    } else {
      #pragma unroll
      for (int e = 0; e < 16; ++e) a[e] = (__bf16)0.0f;
    }
    #pragma unroll
    for (int nt = 0; nt < 8; ++nt) {
      v16bf b = *(const v16bf*)&ldsB[((kc * 8 + nt) * 32 + lane) * 16];
      acc[nt] = __builtin_amdgcn_wmma_f32_16x16x32_bf16(false, a, false, b, (short)0,
                                                        acc[nt], false, false);
    }
  }
  #pragma unroll
  for (int nt = 0; nt < 8; ++nt) {
    int col = nt * 16 + lm;
    #pragma unroll
    for (int r = 0; r < 8; ++r) {
      int m = blockIdx.x * 128 + wv * 16 + half * 8 + r;  // D layout: VGPR r -> M=r(+8 for hi half)
      if (m < n) xs[(size_t)m * HID + col] = acc[nt][r];
    }
  }
}

// ---------- per-node attention coefficients a_s, a_d from xs ----------
__global__ void k_attn_coeff(const float* __restrict__ xs, const float* __restrict__ att_src,
                             const float* __restrict__ att_dst, float* __restrict__ a_s,
                             float* __restrict__ a_d, int n) {
  int idx = blockIdx.x * blockDim.x + threadIdx.x;
  if (idx >= n * HEADS) return;
  int node = idx >> 2, hh = idx & 3;
  const float* r  = xs + (size_t)node * HID + hh * CH;
  const float* ws = att_src + hh * CH;
  const float* wd = att_dst + hh * CH;
  float ss = 0.0f, sd = 0.0f;
  #pragma unroll
  for (int c = 0; c < CH; ++c) { float v = r[c]; ss += v * ws[c]; sd += v * wd[c]; }
  a_s[idx] = ss; a_d[idx] = sd;
}

// ---------- edge pass 1: alpha = leaky_relu(...), segment max via int atomicMax ----------
__global__ void k_edge_max(const int* __restrict__ src, const int* __restrict__ dst,
                           const float* __restrict__ edge_attr, const float* __restrict__ a_s,
                           const float* __restrict__ a_d, const float* __restrict__ ae,
                           float* __restrict__ alpha, int* __restrict__ amax_i, int E_) {
  int e = blockIdx.x * blockDim.x + threadIdx.x;
  if (e >= E_) return;
  int s = src[e], d = dst[e];
  float attr = edge_attr[e];
  #pragma unroll
  for (int hh = 0; hh < HEADS; ++hh) {
    float v = a_s[s * 4 + hh] + a_d[d * 4 + hh] + attr * ae[hh] + ae[4 + hh];
    v = v > 0.0f ? v : 0.2f * v;
    alpha[e * 4 + hh] = v;
    atomicMax(&amax_i[d * 4 + hh], fkey(v));
  }
}

// ---------- edge pass 2: p = exp(alpha - amax[dst]); denom += p (in place over alpha) ----------
__global__ void k_edge_exp(const int* __restrict__ dst, float* __restrict__ alpha,
                           const int* __restrict__ amax_i, float* __restrict__ denom, int E_) {
  int e = blockIdx.x * blockDim.x + threadIdx.x;
  if (e >= E_) return;
  int d = dst[e];
  #pragma unroll
  for (int hh = 0; hh < HEADS; ++hh) {
    float am = fdec(amax_i[d * 4 + hh]);
    float p  = expf(alpha[e * 4 + hh] - am);
    alpha[e * 4 + hh] = p;
    atomicAdd(&denom[d * 4 + hh], p);
  }
}

// ---------- edge pass 3: acc[dst] += xs[src] * attn (one thread per edge-channel) ----------
__global__ void k_edge_scatter(const int* __restrict__ src, const int* __restrict__ dst,
                               const float* __restrict__ alpha, const float* __restrict__ denom,
                               const float* __restrict__ xs, float* __restrict__ acc_,
                               long long total) {
  long long idx = (long long)blockIdx.x * blockDim.x + threadIdx.x;
  if (idx >= total) return;
  int k  = (int)(idx & 127);
  int e  = (int)(idx >> 7);
  int hh = k >> 5;
  int s = src[e], d = dst[e];
  float attn = alpha[e * 4 + hh] / (denom[d * 4 + hh] + 1e-16f);
  atomicAdd(&acc_[(size_t)d * HID + k], xs[(size_t)s * HID + k] * attn);
}

// ---------- bias + batchnorm(eval) + relu + residual, h updated in place ----------
__global__ void k_update(float* __restrict__ h, const float* __restrict__ acc,
                         const float* __restrict__ conv_b, const float* __restrict__ g,
                         const float* __restrict__ be, const float* __restrict__ mu,
                         const float* __restrict__ var, int n) {
  int idx = blockIdx.x * blockDim.x + threadIdx.x;
  if (idx >= n * HID) return;
  int k = idx & 127;
  float o = acc[idx] + conv_b[k];
  o = (o - mu[k]) * rsqrtf(var[k] + 1e-5f) * g[k] + be[k];
  h[idx] = fmaxf(o, 0.0f) + h[idx];
}

// ---------- pooling ----------
__global__ void k_pool_zero(float* __restrict__ pooled, float* __restrict__ cnt, int G_) {
  int idx = blockIdx.x * blockDim.x + threadIdx.x;
  if (idx < G_ * HID) pooled[idx] = 0.0f;
  if (idx < G_) cnt[idx] = 0.0f;
}
__global__ void k_pool(const float* __restrict__ h, const int* __restrict__ batch,
                       float* __restrict__ pooled, float* __restrict__ cnt, int n) {
  int idx = blockIdx.x * blockDim.x + threadIdx.x;
  if (idx >= n * HID) return;
  int node = idx >> 7, k = idx & 127;
  int b = batch[node];
  atomicAdd(&pooled[b * HID + k], h[idx]);
  if (k == 0) atomicAdd(&cnt[b], 1.0f);
}

// ---------- classifier head: out = sigmoid(relu(mean @ w1 + b1) @ w2 + b2) ----------
__global__ void k_classifier(const float* __restrict__ pooled, const float* __restrict__ cnt,
                             const float* __restrict__ w1, const float* __restrict__ b1,
                             const float* __restrict__ w2, const float* __restrict__ b2,
                             float* __restrict__ out, int G_) {
  int g = blockIdx.x * blockDim.x + threadIdx.x;
  if (g >= G_) return;
  float inv = 1.0f / fmaxf(cnt[g], 1.0f);
  float z[HID / 2];
  #pragma unroll
  for (int j = 0; j < HID / 2; ++j) z[j] = b1[j];
  for (int k = 0; k < HID; ++k) {
    float p = pooled[g * HID + k] * inv;
    #pragma unroll
    for (int j = 0; j < HID / 2; ++j) z[j] += p * w1[k * (HID / 2) + j];
  }
  float a = b2[0];
  #pragma unroll
  for (int j = 0; j < HID / 2; ++j) a += fmaxf(z[j], 0.0f) * w2[j];
  out[g] = 1.0f / (1.0f + expf(-a));
}

extern "C" void kernel_launch(void* const* d_in, const int* in_sizes, int n_in,
                              void* d_out, int out_size, void* d_ws, size_t ws_size,
                              hipStream_t stream) {
  const float* x         = (const float*)d_in[0];
  const float* edge_attr = (const float*)d_in[1];
  const int*   edge_idx  = (const int*)d_in[2];
  const int*   batch     = (const int*)d_in[3];
  const float* enc_w     = (const float*)d_in[4];
  const float* enc_b     = (const float*)d_in[5];
  const float* eenc_w    = (const float*)d_in[6];
  const float* eenc_b    = (const float*)d_in[7];
  const float* lin_w     = (const float*)d_in[8];
  const float* att_src   = (const float*)d_in[9];
  const float* att_dst   = (const float*)d_in[10];
  const float* att_edge  = (const float*)d_in[11];
  const float* lin_ew    = (const float*)d_in[12];
  const float* conv_b    = (const float*)d_in[13];
  const float* bn_g      = (const float*)d_in[14];
  const float* bn_b      = (const float*)d_in[15];
  const float* bn_m      = (const float*)d_in[16];
  const float* bn_v      = (const float*)d_in[17];
  const float* w1        = (const float*)d_in[18];
  const float* b1        = (const float*)d_in[19];
  const float* w2        = (const float*)d_in[20];
  const float* b2        = (const float*)d_in[21];

  int N_ = in_sizes[0];
  int E_ = in_sizes[1];
  int L_ = in_sizes[8] / (HID * HID);
  int G_ = out_size;
  const int* src = edge_idx;
  const int* dst = edge_idx + E_;

  char* ws = (char*)d_ws;
  size_t off = 0;
  auto alloc = [&](size_t bytes) { void* p = ws + off; off += (bytes + 255) & ~(size_t)255; return p; };
  float* h      = (float*)alloc((size_t)N_ * HID * 4);
  float* xs     = (float*)alloc((size_t)N_ * HID * 4);
  float* acc    = (float*)alloc((size_t)N_ * HID * 4);
  float* a_s    = (float*)alloc((size_t)N_ * HEADS * 4);
  float* a_d    = (float*)alloc((size_t)N_ * HEADS * 4);
  int*   amax_i = (int*)  alloc((size_t)N_ * HEADS * 4);
  float* denom  = (float*)alloc((size_t)N_ * HEADS * 4);
  float* alpha  = (float*)alloc((size_t)E_ * HEADS * 4);
  float* pooled = (float*)alloc((size_t)G_ * HID * 4);
  float* cnt    = (float*)alloc((size_t)G_ * 4);
  float* ae     = (float*)alloc(64);

  const int tb = 256;
  int NK = N_ * HID;

  k_encode<<<(NK + tb - 1) / tb, tb, 0, stream>>>(x, enc_w, enc_b, h, N_);

  for (int l = 0; l < L_; ++l) {
    k_edge_coeff<<<1, 128, 0, stream>>>(lin_ew + (size_t)l * HID * HID,
                                        att_edge + l * HEADS * CH, eenc_w, eenc_b, ae);
    k_reset<<<(NK + tb - 1) / tb, tb, 0, stream>>>(acc, amax_i, denom, N_);
    k_gemm_xs<<<(N_ + 127) / 128, 256, 0, stream>>>(h, lin_w + (size_t)l * HID * HID, xs, N_);
    k_attn_coeff<<<(N_ * HEADS + tb - 1) / tb, tb, 0, stream>>>(
        xs, att_src + l * HEADS * CH, att_dst + l * HEADS * CH, a_s, a_d, N_);
    k_edge_max<<<(E_ + tb - 1) / tb, tb, 0, stream>>>(src, dst, edge_attr, a_s, a_d, ae,
                                                      alpha, amax_i, E_);
    k_edge_exp<<<(E_ + tb - 1) / tb, tb, 0, stream>>>(dst, alpha, amax_i, denom, E_);
    long long tot = (long long)E_ * HID;
    k_edge_scatter<<<(int)((tot + tb - 1) / tb), tb, 0, stream>>>(src, dst, alpha, denom,
                                                                  xs, acc, tot);
    k_update<<<(NK + tb - 1) / tb, tb, 0, stream>>>(h, acc, conv_b + l * HID, bn_g + l * HID,
                                                    bn_b + l * HID, bn_m + l * HID,
                                                    bn_v + l * HID, N_);
  }

  k_pool_zero<<<(G_ * HID + tb - 1) / tb, tb, 0, stream>>>(pooled, cnt, G_);
  k_pool<<<(NK + tb - 1) / tb, tb, 0, stream>>>(h, batch, pooled, cnt, N_);
  k_classifier<<<1, 64, 0, stream>>>(pooled, cnt, w1, b1, w2, b2, (float*)d_out, G_);
}
